// SimLoss_70222715290004
// MI455X (gfx1250) — matrix operand order repaired and checked by reading, not compiled, not run on hardware
//
#include <hip/hip_runtime.h>

// SimLoss: loss = -sum_ij <e_i/|e_i|, e_j/|e_j|> * sim_x[i,j]
// N = 8192 rows, D = 512 features.

#define NN 8192
#define DD 512

typedef __bf16 bf16_t;
typedef __attribute__((ext_vector_type(8)))  __bf16 v8bf;
typedef __attribute__((ext_vector_type(16))) __bf16 v16bf;
typedef __attribute__((ext_vector_type(8)))  float  v8f;

// Deterministic round-to-nearest-even float -> bf16 (finite inputs only).
__device__ __forceinline__ bf16_t f2bf(float f) {
    union { float f; unsigned u; } v;
    v.f = f;
    unsigned r = v.u + 0x7FFFu + ((v.u >> 16) & 1u);
    unsigned short h = (unsigned short)(r >> 16);
    return __builtin_bit_cast(bf16_t, h);
}

// ---------------------------------------------------------------------------
// Kernel 1: per-row L2 normalize, emit bf16 Ehat into workspace.
// ---------------------------------------------------------------------------
__launch_bounds__(256)
__global__ void simloss_normalize(const float* __restrict__ emb,
                                  bf16_t* __restrict__ ehat) {
    const int row = blockIdx.x;
    const int t   = threadIdx.x;
    const float* r = emb + (size_t)row * DD;
    float x0 = r[t];
    float x1 = r[t + 256];
    __shared__ float red[256];
    red[t] = x0 * x0 + x1 * x1;
    __syncthreads();
    #pragma unroll
    for (int s = 128; s > 0; s >>= 1) {
        if (t < s) red[t] += red[t + s];
        __syncthreads();
    }
    const float inv = rsqrtf(red[0]);
    bf16_t* o = ehat + (size_t)row * DD;
    o[t]       = f2bf(x0 * inv);
    o[t + 256] = f2bf(x1 * inv);
}

// ---------------------------------------------------------------------------
// Kernel 2: block computes a 128x128 gram region contracted against sim_x.
// 8 waves (4 i-strips x 2 j-strips); each wave owns 32x64 (8 v8f accums).
// Per 32-wide K step, 256 threads async-copy A rows (128x32) and B rows
// (128x32) into LDS (double buffered, ASYNCcnt), WMMAs read via ds_load.
// ---------------------------------------------------------------------------
__launch_bounds__(256)
__global__ void simloss_main(const bf16_t* __restrict__ ehat,
                             const float* __restrict__ sim,
                             float* __restrict__ partials) {
    // [buf 0 | buf 1], each buf: A rows 0..127 (32 halves) then B rows 0..127.
    __shared__ __align__(128) bf16_t smem[2 * 8192];   // 32 KB
    __shared__ float wred[8];

    const int t    = threadIdx.x;
    const int lane = t & 31;
    const int wave = t >> 5;
    const int wi   = wave & 3;   // i strip: 32 rows
    const int wj   = wave >> 2;  // j strip: 64 cols
    const int iBase = blockIdx.y * 128;
    const int jBase = blockIdx.x * 128;
    const int lm    = lane & 15;
    const bool hiH  = (lane >= 16);

    // Async staging: thread t owns one 64-byte row slice per K step.
    const int srow = (t < 128) ? (iBase + t) : (jBase + (t - 128));
    const bf16_t* grow = ehat + (size_t)srow * DD;
    const unsigned ldsBase = (unsigned)(size_t)(void*)smem + (unsigned)t * 64u;

    #define STAGE(bufsel, k0)                                                  \
        do {                                                                   \
            unsigned _l = ldsBase + (unsigned)(bufsel) * 16384u;               \
            const bf16_t* _g = grow + (k0);                                    \
            asm volatile(                                                      \
                "global_load_async_to_lds_b128 %0, %1, off\n\t"                \
                "global_load_async_to_lds_b128 %0, %1, off offset:16\n\t"      \
                "global_load_async_to_lds_b128 %0, %1, off offset:32\n\t"      \
                "global_load_async_to_lds_b128 %0, %1, off offset:48"          \
                :: "v"(_l), "v"(_g) : "memory");                               \
        } while (0)

    #define WAIT_ASYNC() asm volatile("s_wait_asynccnt 0x0" ::: "memory")

    v8f acc[2][4] = {};

    const int ka = hiH ? 16 : 0;  // A frag: halves [ka..ka+15]
    const int kb = hiH ? 16 : 0;  // B frag: halves [kb..kb+15]

    STAGE(0, 0);
    WAIT_ASYNC();
    __syncthreads();

    for (int ks = 0; ks < 16; ++ks) {
        if (ks + 1 < 16) STAGE((ks + 1) & 1, (ks + 1) * 32);

        const bf16_t* sb = smem + (ks & 1) * 8192;
        #pragma unroll
        for (int it = 0; it < 2; ++it) {
            const bf16_t* pa = sb + (wi * 32 + it * 16 + lm) * 32;
            v8bf alo = *(const v8bf*)(pa + ka);
            v8bf ahi = *(const v8bf*)(pa + ka + 8);
            v16bf a;
            #pragma unroll
            for (int q = 0; q < 8; ++q) { a[q] = alo[q]; a[q + 8] = ahi[q]; }
            #pragma unroll
            for (int jt = 0; jt < 4; ++jt) {
                const bf16_t* pb = sb + 4096 + (wj * 64 + jt * 16 + lm) * 32 + kb;
                v16bf b = *(const v16bf*)pb;
                acc[it][jt] = __builtin_amdgcn_wmma_f32_16x16x32_bf16(
                    false, a, false, b, (short)0, acc[it][jt], false, false);
            }
        }

        WAIT_ASYNC();       // next-buffer data landed
        __syncthreads();    // all waves done with current buffer
    }

    // C/D layout (16x16 f32): lane -> col n = lane&15; VGPR r -> row r + 8*(lane>>4).
    // Stream sim_x non-temporally so Ehat stays resident in L2.
    float psum = 0.0f;
    #pragma unroll
    for (int it = 0; it < 2; ++it) {
        const int mb = iBase + wi * 32 + it * 16 + (hiH ? 8 : 0);
        #pragma unroll
        for (int jt = 0; jt < 4; ++jt) {
            const int j = jBase + wj * 64 + jt * 16 + lm;
            #pragma unroll
            for (int r = 0; r < 8; ++r) {
                psum += acc[it][jt][r] *
                        __builtin_nontemporal_load(sim + (size_t)(mb + r) * NN + j);
            }
        }
    }

    #pragma unroll
    for (int off = 16; off > 0; off >>= 1)
        psum += __shfl_xor(psum, off, 32);

    if (lane == 0) wred[wave] = psum;
    __syncthreads();
    if (t == 0) {
        float s = 0.0f;
        #pragma unroll
        for (int w = 0; w < 8; ++w) s += wred[w];
        partials[blockIdx.y * gridDim.x + blockIdx.x] = s;
    }
}

// ---------------------------------------------------------------------------
// Kernel 3: deterministic final reduction of block partials, negate.
// ---------------------------------------------------------------------------
__launch_bounds__(256)
__global__ void simloss_reduce(const float* __restrict__ partials,
                               float* __restrict__ out, int n) {
    const int t = threadIdx.x;
    float s = 0.0f;
    for (int i = t; i < n; i += 256) s += partials[i];
    __shared__ float red[256];
    red[t] = s;
    __syncthreads();
    #pragma unroll
    for (int k = 128; k > 0; k >>= 1) {
        if (t < k) red[t] += red[t + k];
        __syncthreads();
    }
    if (t == 0) out[0] = -red[0];
}

extern "C" void kernel_launch(void* const* d_in, const int* in_sizes, int n_in,
                              void* d_out, int out_size, void* d_ws, size_t ws_size,
                              hipStream_t stream) {
    const float* emb = (const float*)d_in[0];   // [8192, 512] fp32
    const float* sim = (const float*)d_in[1];   // [8192, 8192] fp32
    // d_in[2] (beta) is unused by the reference computation.

    bf16_t* ehat     = (bf16_t*)d_ws;                               // 8 MB
    float*  partials = (float*)((char*)d_ws + (size_t)NN * DD * 2); // 16 KB

    simloss_normalize<<<NN, 256, 0, stream>>>(emb, ehat);

    dim3 grid(NN / 128, NN / 128); // (64, 64)
    simloss_main<<<grid, 256, 0, stream>>>(ehat, sim, partials);

    simloss_reduce<<<1, 256, 0, stream>>>(partials, (float*)d_out,
                                          (NN / 128) * (NN / 128));
}
